// SALayer_11768210391132
// MI455X (gfx1250) — matrix operand order repaired and checked by reading, not compiled, not run on hardware
//
#include <hip/hip_runtime.h>
#include <hip/hip_bf16.h>

// ---------------------------------------------------------------------------
// SALayer (multi-head self attention, 1x1-conv projections)
// MI455X / gfx1250: wave32, WMMA f32_16x16x32_f16, flash-attention fusion.
// All LDS operand tiles stored K-contiguous so WMMA fragments load as
// contiguous b128 chunks (no scalar u16 gathers).
// ---------------------------------------------------------------------------

typedef __attribute__((ext_vector_type(16))) _Float16 v16h;
typedef __attribute__((ext_vector_type(8)))  _Float16 v8h;
typedef __attribute__((ext_vector_type(4)))  _Float16 v4h;
typedef __attribute__((ext_vector_type(8)))  float    v8f;

#define IN_DIM  512
#define ATT_DIM 256
#define HEADS   8
#define BSZ     8
#define NPOS    2048
#define DK      32      // per-head key/query dim  (256/8)
#define DVD     64      // per-head value dim      (512/8)
#define SCALE_F 0.0625f // 1/sqrt(256)

// Concatenate two v8h halves into a v16h A/B fragment register set.
__device__ __forceinline__ v16h cat16(v8h lo, v8h hi) {
  return __builtin_shufflevector(lo, hi, 0, 1, 2, 3, 4, 5, 6, 7, 8, 9, 10, 11,
                                 12, 13, 14, 15);
}

__device__ __forceinline__ v8f zero8() {
  v8f z;
#pragma unroll
  for (int i = 0; i < 8; ++i) z[i] = 0.0f;
  return z;
}

__device__ __forceinline__ v8f wmma16(v16h a, v16h b, v8f c) {
  // D = A(16x32 f16) * B(32x16 f16) + C(16x16 f32)
  return __builtin_amdgcn_wmma_f32_16x16x32_f16(false, a, false, b,
                                                (short)0, c, false, false);
}

// Fragment loaders.  A tile in LDS is [M][K] row-major (stride in f16);
// A K-map is {16lh + e} (same as B): two contiguous 16B chunks.
__device__ __forceinline__ v16h loadA(const _Float16* row, int lh) {
  v8h lo = *(const v8h*)(row + 16 * lh);
  v8h hi = *(const v8h*)(row + 16 * lh + 8);
  return cat16(lo, hi);
}
// B tile in LDS is [N][K]; B K-map is {16lh + e}: one 32B chunk.
__device__ __forceinline__ v16h loadB(const _Float16* row, int lh) {
  return *(const v16h*)(row + 16 * lh);
}

// ---------------------------------------------------------------------------
// Kernel 1: projection GEMM  out[o,n] = sum_c W[o,c]*feats[b,c,n] + bias[o]
// NMAJOR=true : out f16 [b][h][n][Dh]   (for K and Q; Q pre-scaled)
// NMAJOR=false: out f16 [b][h][Dh][n]   (for V)
// Head split follows reshape(bs,-1,8,n): h = o%8, d = o/8.
// Block: 128 threads (4 waves); tile 64(M) x 64(N); K-step 32.
// ---------------------------------------------------------------------------
template <bool NMAJOR>
__global__ __launch_bounds__(128) void sal_proj_kernel(
    const float* __restrict__ feats, const float* __restrict__ W,
    const float* __restrict__ bias, _Float16* __restrict__ out, int Dh,
    float scale) {
  __shared__ alignas(16) _Float16 Wsh[64][40];  // [m][k], row-major, pad 8
  __shared__ alignas(16) _Float16 Xsh[64][40];  // [n][k], transposed feats

  const int tid = threadIdx.x;
  const int w = tid >> 5, lane = tid & 31;
  const int lrow = lane & 15, lh = lane >> 4;
  const int b = blockIdx.z;
  const int oBase = blockIdx.y * 64;
  const int nBase = blockIdx.x * 64;

  v8f acc[4];
#pragma unroll
  for (int t = 0; t < 4; ++t) acc[t] = zero8();

  for (int k0 = 0; k0 < IN_DIM; k0 += 32) {
    __syncthreads();
    // Stage W 64x32: float4 global loads, packed v4h LDS stores (row-major).
#pragma unroll
    for (int i = 0; i < 4; ++i) {
      int idx = tid + i * 128;  // 512 float4
      int row = idx >> 3, c4 = idx & 7;
      float4 f =
          *(const float4*)&W[(size_t)(oBase + row) * IN_DIM + k0 + c4 * 4];
      v4h p;
      p[0] = (_Float16)f.x; p[1] = (_Float16)f.y;
      p[2] = (_Float16)f.z; p[3] = (_Float16)f.w;
      *(v4h*)&Wsh[row][c4 * 4] = p;
    }
    // Stage feats 32(k)x64(n), transposed into Xsh[n][k].
#pragma unroll
    for (int i = 0; i < 4; ++i) {
      int idx = tid + i * 128;  // 512 float4
      int row = idx >> 4, c4 = idx & 15;
      float4 f = *(const float4*)&feats[((size_t)b * IN_DIM + k0 + row) * NPOS +
                                        nBase + c4 * 4];
      Xsh[c4 * 4 + 0][row] = (_Float16)f.x;
      Xsh[c4 * 4 + 1][row] = (_Float16)f.y;
      Xsh[c4 * 4 + 2][row] = (_Float16)f.z;
      Xsh[c4 * 4 + 3][row] = (_Float16)f.w;
    }
    __syncthreads();

    v16h a = loadA(&Wsh[w * 16 + lrow][0], lh);
#pragma unroll
    for (int nt = 0; nt < 4; ++nt) {
      v16h bf = loadB(&Xsh[nt * 16 + lrow][0], lh);
      acc[nt] = wmma16(a, bf, acc[nt]);
    }
  }

  // Bias + store (C/D layout: M = 8*lh + r, N = lrow).
  float bv[8];
#pragma unroll
  for (int r = 0; r < 8; ++r) bv[r] = bias[oBase + w * 16 + 8 * lh + r];
#pragma unroll
  for (int nt = 0; nt < 4; ++nt) {
#pragma unroll
    for (int r = 0; r < 8; ++r) {
      int o = oBase + w * 16 + 8 * lh + r;
      int h = o & (HEADS - 1), d = o >> 3;
      int n = nBase + nt * 16 + lrow;
      float v = (acc[nt][r] + bv[r]) * scale;
      size_t idx = NMAJOR ? (((size_t)b * HEADS + h) * NPOS + n) * Dh + d
                          : (((size_t)b * HEADS + h) * Dh + d) * NPOS + n;
      out[idx] = (_Float16)v;
    }
  }
}

// ---------------------------------------------------------------------------
// Kernel 2: flash attention per (b, h).  Block: 4 waves x 16 queries = 64 q.
// Per 64-key tile: S = K^T Q (4 WMMA), online softmax, O += V*P (8 WMMA).
// ---------------------------------------------------------------------------
__global__ __launch_bounds__(128) void sal_attn_kernel(
    const _Float16* __restrict__ Kbuf,  // [B][H][N][DK]  (n-major)
    const _Float16* __restrict__ Qbuf,  // [B][H][N][DK]  (n-major, pre-scaled)
    const _Float16* __restrict__ Vbuf,  // [B][H][DVD][N] (dv-major)
    const float* __restrict__ feats, const float* __restrict__ masks,
    float* __restrict__ out) {
  __shared__ alignas(16) _Float16 Ksh[64][40];      // [key][d]
  __shared__ alignas(16) _Float16 Vsh[DVD][72];     // [dv][key]
  __shared__ alignas(16) _Float16 Psh[4][16][72];   // per-wave [q][key]
  __shared__ float Pen[64];

  const int tid = threadIdx.x;
  const int w = tid >> 5, lane = tid & 31;
  const int lrow = lane & 15, lh = lane >> 4;
  const int b = blockIdx.z, h = blockIdx.y;
  const int qBase = blockIdx.x * 64 + w * 16;

  const size_t kqBase = ((size_t)b * HEADS + h) * NPOS;  // rows of [N][DK]
  const size_t vOff = ((size_t)b * HEADS + h) * DVD * NPOS;

  // Q fragment (B operand): one contiguous 32B global load, lives in regs.
  v16h qf = *(const v16h*)&Qbuf[(kqBase + qBase + lrow) * DK + 16 * lh];

  v8f o_acc[4];
#pragma unroll
  for (int t = 0; t < 4; ++t) o_acc[t] = zero8();
  float m = -1e30f, lsum = 0.0f;

  for (int k0 = 0; k0 < NPOS; k0 += 64) {
    __syncthreads();  // protect LDS re-staging vs previous iteration reads
    {
      // K tile: 64 rows x 32 f16 = one contiguous 4KB block (n-major layout).
      const uint4* ks = (const uint4*)(Kbuf + (kqBase + k0) * DK);
#pragma unroll
      for (int i = 0; i < 2; ++i) {
        int idx = tid + i * 128;  // 256 uint4
        int row = idx >> 2, c4 = idx & 3;
        *(uint4*)&Ksh[row][c4 * 8] = ks[idx];
      }
      // V tile: 64 rows x 64 f16, row stride NPOS.
      const uint4* vs = (const uint4*)(Vbuf + vOff + k0);
#pragma unroll
      for (int i = 0; i < 4; ++i) {
        int idx = tid + i * 128;  // 512 uint4
        int row = idx >> 3, c8 = idx & 7;
        *(uint4*)&Vsh[row][c8 * 8] = vs[(size_t)row * (NPOS / 8) + c8];
      }
      if (tid < 64)
        Pen[tid] = (1.0f - masks[(size_t)b * NPOS + k0 + tid]) * 1e8f;
      if (k0 + 64 < NPOS)  // hint next K tile toward the caches
        __builtin_prefetch(Kbuf + (kqBase + k0 + 64) * DK, 0, 1);
    }
    __syncthreads();

    // S[k,q] = sum_d K[k][d] * Q[q][d]   (A = Ksh rows, Kdim = 32)
    v8f s[4];
#pragma unroll
    for (int mt = 0; mt < 4; ++mt) {
      v16h a = loadA(&Ksh[mt * 16 + lrow][0], lh);
      s[mt] = wmma16(a, qf, zero8());
    }

    // Online softmax over keys. Column q lives in lanes {q, q+16}.
    float tmax = -1e30f;
#pragma unroll
    for (int mt = 0; mt < 4; ++mt)
#pragma unroll
      for (int r = 0; r < 8; ++r) {
        float sv = s[mt][r] - Pen[mt * 16 + 8 * lh + r];
        s[mt][r] = sv;
        tmax = fmaxf(tmax, sv);
      }
    tmax = fmaxf(tmax, __shfl_xor(tmax, 16, 32));
    float mnew = fmaxf(m, tmax);
    float corr = __expf(m - mnew);
    float psum = 0.0f;
    // P stored [q][key]: each S tile gives 8 consecutive keys -> one 16B store.
#pragma unroll
    for (int mt = 0; mt < 4; ++mt) {
      v8h pv;
#pragma unroll
      for (int r = 0; r < 8; ++r) {
        float p = __expf(s[mt][r] - mnew);
        psum += p;
        pv[r] = (_Float16)p;
      }
      *(v8h*)&Psh[w][lrow][mt * 16 + 8 * lh] = pv;
    }
    psum += __shfl_xor(psum, 16, 32);
    lsum = lsum * corr + psum;
    m = mnew;
#pragma unroll
    for (int t = 0; t < 4; ++t)
#pragma unroll
      for (int r = 0; r < 8; ++r) o_acc[t][r] *= corr;

    __syncthreads();  // Psh visible; keeps waves in lockstep with staging

    // O[dv,q] += sum_k V[dv][k] * P[q][k]^T  (Kdim = 64 -> two K-steps of 32)
#pragma unroll
    for (int kh = 0; kh < 2; ++kh) {
      v16h pf = loadB(&Psh[w][lrow][kh * 32], lh);
#pragma unroll
      for (int dvt = 0; dvt < 4; ++dvt) {
        v16h a = loadA(&Vsh[dvt * 16 + lrow][kh * 32], lh);
        o_acc[dvt] = wmma16(a, pf, o_acc[dvt]);
      }
    }
  }

  // Normalize, re-interleave heads (c = dv*8 + h), add residual, store fp32.
  float inv = (lsum > 0.0f) ? 1.0f / lsum : 0.0f;
  int q = qBase + lrow;
#pragma unroll
  for (int dvt = 0; dvt < 4; ++dvt)
#pragma unroll
    for (int r = 0; r < 8; ++r) {
      int dv = dvt * 16 + 8 * lh + r;
      int c = dv * HEADS + h;
      size_t idx = ((size_t)b * IN_DIM + c) * NPOS + q;
      out[idx] = o_acc[dvt][r] * inv + feats[idx];
    }
}

// ---------------------------------------------------------------------------
extern "C" void kernel_launch(void* const* d_in, const int* in_sizes, int n_in,
                              void* d_out, int out_size, void* d_ws,
                              size_t ws_size, hipStream_t stream) {
  const float* feats = (const float*)d_in[0];
  const float* masks = (const float*)d_in[1];
  const float* Wk = (const float*)d_in[2];
  const float* bk = (const float*)d_in[3];
  const float* Wq = (const float*)d_in[4];
  const float* bq = (const float*)d_in[5];
  const float* Wv = (const float*)d_in[6];
  const float* bv = (const float*)d_in[7];
  float* out = (float*)d_out;

  _Float16* Kbuf = (_Float16*)d_ws;                         // [8][8][2048][32]
  _Float16* Qbuf = Kbuf + (size_t)BSZ * HEADS * NPOS * DK;  // [8][8][2048][32]
  _Float16* Vbuf = Qbuf + (size_t)BSZ * HEADS * NPOS * DK;  // [8][8][64][2048]

  dim3 blk(128, 1, 1);
  sal_proj_kernel<true><<<dim3(32, 4, 8), blk, 0, stream>>>(feats, Wk, bk,
                                                            Kbuf, DK, 1.0f);
  sal_proj_kernel<true><<<dim3(32, 4, 8), blk, 0, stream>>>(feats, Wq, bq,
                                                            Qbuf, DK, SCALE_F);
  sal_proj_kernel<false><<<dim3(32, 8, 8), blk, 0, stream>>>(feats, Wv, bv,
                                                             Vbuf, DVD, 1.0f);
  sal_attn_kernel<<<dim3(32, 8, 8), blk, 0, stream>>>(Kbuf, Qbuf, Vbuf, feats,
                                                      masks, out);
}
